// DIFFormerConv_73572789781363
// MI455X (gfx1250) — compile-verified, matrix operand order checked by reading, not run on hardware
//
#include <hip/hip_runtime.h>
#include <hip/hip_bf16.h>
#include <stdint.h>

typedef __attribute__((ext_vector_type(16))) __bf16 v16bf;
typedef __attribute__((ext_vector_type(8)))  __bf16 v8bf;
typedef __attribute__((ext_vector_type(8)))  float  v8f;
typedef __attribute__((ext_vector_type(4)))  unsigned int v4u;
typedef __attribute__((ext_vector_type(8)))  int    v8i;
typedef __attribute__((ext_vector_type(4)))  int    v4i;

constexpr int GN  = 65536;  // nodes
constexpr int INC = 512;    // in channels
constexpr int GD  = 256;    // out channels per head
constexpr int GH  = 4;      // heads
constexpr int HD  = 1024;   // GH*GD

// ---------------------------------------------------------------- TDM support
#if defined(__AMDGCN__) && __has_builtin(__builtin_amdgcn_tensor_load_to_lds) && \
    __has_builtin(__builtin_amdgcn_s_wait_tensorcnt)
#define USE_TDM 1
#else
#define USE_TDM 0
#endif

__device__ __forceinline__ uint32_t lds_off(const void* p) {
  return (uint32_t)(uintptr_t)p;  // low 32 bits of flat LDS addr = LDS byte address
}

#if USE_TDM
// 2D tile load via Tensor Data Mover (D# layout per CDNA5 ISA).
// dsz: 0=1B,1=2B,2=4B. Optional LDS padding: pad_amt+1 DWORDs every
// (2<<pad_int) DWORDs.
__device__ __forceinline__ void tdm_load_2d(uint32_t lds_addr, const void* gptr,
                                            uint32_t tile_w, uint32_t tile_h,
                                            uint32_t pitch_elems, uint32_t dsz,
                                            uint32_t pad_en, uint32_t pad_int,
                                            uint32_t pad_amt) {
  const uint64_t ga = (uint64_t)(uintptr_t)gptr;
  v4u g0;
  g0[0] = 1u;                                   // count=1 (valid user descriptor)
  g0[1] = lds_addr;                             // lds_addr (bytes)
  g0[2] = (uint32_t)ga;                         // global_addr[31:0]
  g0[3] = (uint32_t)(ga >> 32) | (2u << 30);    // global_addr[56:32] | type=2
  v8i g1;
  g1[0] = (int)((dsz << 16) | (pad_en << 20) | (pad_int << 22) | (pad_amt << 25));
  g1[1] = (int)((tile_w & 0xFFFFu) << 16);      // tensor_dim0[15:0]
  g1[2] = (int)(((tile_w >> 16) & 0xFFFFu) | ((tile_h & 0xFFFFu) << 16));
  g1[3] = (int)(((tile_h >> 16) & 0xFFFFu) | ((tile_w & 0xFFFFu) << 16)); // tile_dim0
  g1[4] = (int)(tile_h & 0xFFFFu);              // tile_dim1 (tile_dim2=0)
  g1[5] = (int)pitch_elems;                     // tensor_dim0_stride[31:0]
  g1[6] = 0;
  g1[7] = 0;
#if __clang_major__ >= 23
  __builtin_amdgcn_tensor_load_to_lds(g0, g1, (v4i)0, (v4i)0, (v8i)0, 0);
#else
  __builtin_amdgcn_tensor_load_to_lds(g0, g1, (v4i)0, (v4i)0, 0);
#endif
}
#endif

// ------------------------------------------------------------- WMMA fragments
// K-index mapping for 16-bit WMMA fragments: element j of a lane covers
// k = 8*kh + j (j<8) and k = 16 + 8*kh + (j-8) (j>=8).
__device__ __forceinline__ int kidx(int j, int kh) {
  return (j < 8) ? (kh * 8 + j) : (16 + kh * 8 + (j - 8));
}

// Fragment = two contiguous 8-element bf16 runs at p and p+16, loaded through
// an explicit global (AS1) pointer so they lower to global_load_b128 (not flat).
typedef const __attribute__((address_space(1))) v8bf* gv8bf_p;

__device__ __forceinline__ v16bf frag2x8(const __bf16* p) {
  gv8bf_p gp = (gv8bf_p)(uintptr_t)p;
  v8bf lo = gp[0];
  v8bf hi = gp[2];  // +16 elements
  v16bf f;
#pragma unroll
  for (int j = 0; j < 8; ++j) { f[j] = lo[j]; f[8 + j] = hi[j]; }
  return f;
}

__device__ __forceinline__ v8f wmma_bf16(v16bf a, v16bf b, v8f c) {
  return __builtin_amdgcn_wmma_f32_16x16x32_bf16(false, a, false, b, (short)0, c,
                                                 false, false);
}

// Packed B layout: logical [K x C] matrix in 32(K)x16(C) tiles of 512 elems;
// tile index (c>>4)*(K/32) + (k>>5); element (c&15)*32 + (k&31).
__device__ __forceinline__ size_t pk_base(int k, int c, int Kdim) {
  return ((size_t)(c >> 4) * (Kdim >> 5) + (k >> 5)) * 512 + ((c & 15) << 5);
}

// ---------------------------------------------------------------- prep kernels
__global__ __launch_bounds__(256) void k_zero(float* __restrict__ p, int n) {
  int i = blockIdx.x * 256 + threadIdx.x;
  if (i < n) p[i] = 0.0f;
}

__global__ __launch_bounds__(256)
void k_cvt(const float* __restrict__ src, __bf16* __restrict__ dst) {
  const size_t i = (size_t)blockIdx.x * 256 + threadIdx.x;
  dst[i] = (__bf16)src[i];
}

// W (f32 row-major [512][1024] x3) -> packed bf16 (one-time, L2 resident)
__global__ __launch_bounds__(256)
void k_cvt_w_pack(const float* __restrict__ Wq, const float* __restrict__ Wk,
                  const float* __restrict__ Wv, __bf16* __restrict__ Wbp) {
  const int chunk = blockIdx.x * 256 + threadIdx.x;  // 3*65536 chunks of 8
  const int sel = chunk >> 16;
  int u = chunk & 65535;
  const int k8 = u & 3;  u >>= 2;
  const int cl = u & 15; u >>= 4;
  const int kt = u & 15; u >>= 4;
  const int ct = u;  // 0..63
  const float* S = (sel == 0) ? Wq : (sel == 1) ? Wk : Wv;
  const int c  = ct * 16 + cl;
  const int k0 = kt * 32 + k8 * 8;
  v8bf o;
#pragma unroll
  for (int i = 0; i < 8; ++i) o[i] = (__bf16)S[(size_t)(k0 + i) * HD + c];
  *(v8bf*)(Wbp + (size_t)sel * (INC * HD) + pk_base(k0, c, INC) + k8 * 8) = o;
}

// kv (f32 [4][256][256]) -> packed bf16 kvp
__global__ __launch_bounds__(256)
void k_cvt_kv_pack(const float* __restrict__ kv, __bf16* __restrict__ kvp) {
  const int chunk = blockIdx.x * 256 + threadIdx.x;  // 32768 chunks of 8
  const int h = chunk >> 13;
  int u = chunk & 8191;
  const int k8 = u & 3;  u >>= 2;
  const int cl = u & 15; u >>= 4;
  const int kt = u & 7;  u >>= 3;
  const int ct = u;  // 0..15
  const int c  = ct * 16 + cl;
  const int k0 = kt * 32 + k8 * 8;
  v8bf o;
#pragma unroll
  for (int i = 0; i < 8; ++i) o[i] = (__bf16)kv[(h << 16) + ((k0 + i) << 8) + c];
  *(v8bf*)(kvp + (size_t)h * (GD * GD) + pk_base(k0, c, GD) + k8 * 8) = o;
}

// -------------------------------------------- QKV projection (bf16 WMMA GEMM)
// grid (48, N/64): pure global b128 frag loads (A row-major xb, B packed Wbp).
__global__ __launch_bounds__(128)
void k_proj(const __bf16* __restrict__ xb, const __bf16* __restrict__ Wbp,
            const float* __restrict__ bq, const float* __restrict__ bk,
            const float* __restrict__ bv,
            __bf16* __restrict__ Qb, __bf16* __restrict__ Kb,
            __bf16* __restrict__ Vb,
            float* __restrict__ sqq, float* __restrict__ sqk) {
  const int lane = threadIdx.x & 31;
  const int wave = threadIdx.x >> 5;
  const int kh   = lane >> 4;
  const int ln   = lane & 15;

  const int cb   = blockIdx.x;      // 0..47
  const int sel  = cb >> 4;         // 0=q 1=k 2=v
  const int colb = (cb & 15) * 64;  // column base within 1024
  const int row0 = blockIdx.y * 64 + wave * 16;

  const float* bias = (sel == 0) ? bq : (sel == 1) ? bk : bv;
  __bf16*      Out  = (sel == 0) ? Qb : (sel == 1) ? Kb : Vb;

  v8f acc[4];
#pragma unroll
  for (int t = 0; t < 4; ++t) acc[t] = (v8f)0.0f;

  const __bf16* ap = xb + (size_t)(row0 + ln) * INC + 8 * kh;
  const __bf16* bp[4];
#pragma unroll
  for (int t = 0; t < 4; ++t)
    bp[t] = Wbp + (size_t)sel * (INC * HD) + pk_base(0, colb + t * 16 + ln, INC) + 8 * kh;

  for (int kk = 0; kk < INC; kk += 32) {
    // load phase: all fragments of this K-step
    v16bf a = frag2x8(ap + kk);
    v16bf b[4];
#pragma unroll
    for (int t = 0; t < 4; ++t) {
      b[t] = frag2x8(bp[t]);
      bp[t] += 512;
    }
    // math phase
#pragma unroll
    for (int t = 0; t < 4; ++t) acc[t] = wmma_bf16(a, b[t], acc[t]);
  }

#pragma unroll
  for (int r = 0; r < 8; ++r) {
    const int row = row0 + r + 8 * kh;
    float s = 0.0f;
#pragma unroll
    for (int t = 0; t < 4; ++t) {
      const int col = colb + t * 16 + ln;
      float v = acc[t][r] + bias[col];
      Out[(size_t)row * HD + col] = (__bf16)v;
      s += v * v;
    }
    if (sel < 2) {
#pragma unroll
      for (int off = 8; off > 0; off >>= 1) s += __shfl_xor(s, off, 16);
      if (ln == 0) {
        float* sq = (sel == 0) ? sqq : sqk;
        atomicAdd(&sq[row * GH + (colb >> 8)], s);
      }
    }
  }
}

// ------------------------------------------ in-place L2 normalization of Q, K
__global__ __launch_bounds__(256)
void k_norm(__bf16* __restrict__ Qb, __bf16* __restrict__ Kb,
            const float* __restrict__ sqq, const float* __restrict__ sqk) {
  const size_t idx = (size_t)blockIdx.x * 256 + threadIdx.x;
  const int n = (int)(idx >> 10);
  const int h = (int)((idx & 1023) >> 8);
  Qb[idx] = (__bf16)((float)Qb[idx] * rsqrtf(sqq[n * GH + h]));
  Kb[idx] = (__bf16)((float)Kb[idx] * rsqrtf(sqk[n * GH + h]));
}

// ----------------- transpose row-major [N][1024] bf16 into packed K-major tiles
// grid (N/64, 16). 64x64 tile staged in LDS by TDM (pitch padded to 72 via the
// descriptor's pad fields); transposed b128 writes to the packed buffer.
__global__ __launch_bounds__(256)
void k_pack(const __bf16* __restrict__ Src, __bf16* __restrict__ Dst) {
  __shared__ __align__(16) __bf16 sT[64 * 72];
  const int tid = threadIdx.x;
  const int n0  = blockIdx.x * 64;
  const int c0  = blockIdx.y * 64;
#if USE_TDM
  if (tid < 32) {  // wave 0 issues the DMA and waits on TENSORcnt
    tdm_load_2d(lds_off(sT), Src + (size_t)n0 * HD + c0, 64, 64, HD,
                /*dsz 2B*/1, /*pad*/1, /*every 32 dwords*/4, /*4 dwords*/3);
    __builtin_amdgcn_s_wait_tensorcnt(0);
  }
#else
#pragma unroll
  for (int i = 0; i < 2; ++i) {
    const int u = tid + i * 256;
    const int r = u >> 3, q = u & 7;
    ((uint4*)(sT + r * 72))[q] = ((const uint4*)(Src + (size_t)(n0 + r) * HD + c0))[q];
  }
#endif
  __syncthreads();
#pragma unroll
  for (int i = 0; i < 2; ++i) {
    int u = tid + i * 256;  // 512 chunks of 8
    const int k8 = u & 3;  u >>= 2;
    const int cl = u & 15; u >>= 4;
    const int kt = u & 1;  u >>= 1;
    const int ct = u;  // 0..3
    v8bf o;
#pragma unroll
    for (int j = 0; j < 8; ++j) o[j] = sT[(kt * 32 + k8 * 8 + j) * 72 + ct * 16 + cl];
    const int c = c0 + ct * 16 + cl;
    const int k = n0 + kt * 32;
    *(v8bf*)(Dst + pk_base(k, c, GN) + k8 * 8) = o;
  }
}

// -------------------------------- kv[h] = Kn^T @ V, split-K over N (atomics)
// grid (16, 16): all fragments are 2x global b128 loads from packed Ktp/Vtp.
__global__ __launch_bounds__(256)
void k_kv(const __bf16* __restrict__ Ktp, const __bf16* __restrict__ Vtp,
          float* __restrict__ kv) {
  const int lane = threadIdx.x & 31;
  const int wave = threadIdx.x >> 5;
  const int kh   = lane >> 4;
  const int ln   = lane & 15;

  const int h     = blockIdx.x & 3;
  const int mslab = blockIdx.x >> 2;        // 0..3 (64 kv rows each)
  const int n0    = blockIdx.y * 4096;
  const int mloc  = (wave & 3) * 16;        // local dK row strip
  const int colh  = (wave >> 2) * 128;      // dV col base

  const int cA = h * GD + mslab * 64 + mloc + ln;
  const __bf16* ap = Ktp + pk_base(n0, cA, GN) + 8 * kh;
  const __bf16* bp[8];
#pragma unroll
  for (int t = 0; t < 8; ++t)
    bp[t] = Vtp + pk_base(n0, h * GD + colh + t * 16 + ln, GN) + 8 * kh;

  v8f acc[8];
#pragma unroll
  for (int t = 0; t < 8; ++t) acc[t] = (v8f)0.0f;

  for (int kk = 0; kk < 4096; kk += 32) {
    // load phase
    v16bf a = frag2x8(ap);
    ap += 512;
    v16bf b[8];
#pragma unroll
    for (int t = 0; t < 8; ++t) {
      b[t] = frag2x8(bp[t]);
      bp[t] += 512;
    }
    // math phase
#pragma unroll
    for (int t = 0; t < 8; ++t) acc[t] = wmma_bf16(a, b[t], acc[t]);
  }

#pragma unroll
  for (int t = 0; t < 8; ++t) {
#pragma unroll
    for (int r = 0; r < 8; ++r) {
      const int m = mslab * 64 + mloc + r + 8 * kh;
      const int c = colh + t * 16 + ln;
      atomicAdd(&kv[(h << 16) + (m << 8) + c], acc[t][r]);
    }
  }
}

// ------------------------------------------- ksum = sum_n kn, vsum = sum_n v
__global__ __launch_bounds__(256)
void k_sums(const __bf16* __restrict__ Kb, const __bf16* __restrict__ Vb,
            float* __restrict__ ksum, float* __restrict__ vsum) {
  const int n0 = blockIdx.x * 64;
  const int t  = threadIdx.x;
  float ks[4] = {0, 0, 0, 0}, vs[4] = {0, 0, 0, 0};
  for (int r = 0; r < 64; ++r) {
    const size_t base = (size_t)(n0 + r) * HD;
#pragma unroll
    for (int i = 0; i < 4; ++i) {
      const int c = t + i * 256;
      ks[i] += (float)Kb[base + c];
      vs[i] += (float)Vb[base + c];
    }
  }
#pragma unroll
  for (int i = 0; i < 4; ++i) {
    atomicAdd(&ksum[t + i * 256], ks[i]);
    atomicAdd(&vsum[t + i * 256], vs[i]);
  }
}

// ---------------- out[n,d] = mean_h (qn@kv[h] + vsum[h]) / (qn.ksum[h] + N)
// grid N/64; 8 waves cover a 64x256 tile; B frags from packed kvp.
__global__ __launch_bounds__(256)
void k_out(const __bf16* __restrict__ Qb, const __bf16* __restrict__ kvp,
           const float* __restrict__ ksum, const float* __restrict__ vsum,
           float* __restrict__ out) {
  __shared__ __align__(16) float s_ksum[HD];
  __shared__ __align__(16) float s_vsum[HD];
#if USE_TDM
  if (threadIdx.x < 32) {
    tdm_load_2d(lds_off(s_ksum), ksum, HD, 1, HD, /*dsz 4B*/2, 0, 0, 0);
    tdm_load_2d(lds_off(s_vsum), vsum, HD, 1, HD, /*dsz 4B*/2, 0, 0, 0);
    __builtin_amdgcn_s_wait_tensorcnt(0);
  }
#else
  for (int i = threadIdx.x; i < HD; i += 256) {
    s_ksum[i] = ksum[i];
    s_vsum[i] = vsum[i];
  }
#endif
  __syncthreads();

  const int lane = threadIdx.x & 31;
  const int wave = threadIdx.x >> 5;
  const int kh   = lane >> 4;
  const int ln   = lane & 15;

  const int n0     = blockIdx.x * 64;
  const int rstrip = (wave & 3) * 16;
  const int colh   = (wave >> 2) * 128;

  const __bf16* arow = Qb + (size_t)(n0 + rstrip + ln) * HD + 8 * kh;

  v8f oacc[8];
#pragma unroll
  for (int t = 0; t < 8; ++t) oacc[t] = (v8f)0.0f;

  for (int h = 0; h < GH; ++h) {
    v8f acc[8];
#pragma unroll
    for (int t = 0; t < 8; ++t) acc[t] = (v8f)0.0f;
    float dot = 0.0f;

    const __bf16* bp[8];
#pragma unroll
    for (int t = 0; t < 8; ++t)
      bp[t] = kvp + (size_t)h * (GD * GD) + pk_base(0, colh + t * 16 + ln, GD) + 8 * kh;

    for (int kk = 0; kk < GD; kk += 32) {
      // load phase
      v16bf a = frag2x8(arow + h * GD + kk);
      v16bf b[8];
#pragma unroll
      for (int t = 0; t < 8; ++t) {
        b[t] = frag2x8(bp[t]);
        bp[t] += 512;
      }
      // math phase (+ denominator GEMV on the A fragment)
#pragma unroll
      for (int j = 0; j < 16; ++j)
        dot += (float)a[j] * s_ksum[h * GD + kk + kidx(j, kh)];
#pragma unroll
      for (int t = 0; t < 8; ++t) acc[t] = wmma_bf16(a, b[t], acc[t]);
    }
    dot += __shfl_xor(dot, 16, 32);  // full qn.ksum per row (row = ln)

#pragma unroll
    for (int r = 0; r < 8; ++r) {
      const float dn = __shfl(dot, r + 8 * kh, 32) + 65536.0f;  // + N
#pragma unroll
      for (int t = 0; t < 8; ++t) {
        const int c = colh + t * 16 + ln;
        oacc[t][r] += 0.25f * (acc[t][r] + s_vsum[h * GD + c]) / dn;
      }
    }
  }

#pragma unroll
  for (int t = 0; t < 8; ++t) {
#pragma unroll
    for (int r = 0; r < 8; ++r) {
      const int c = colh + t * 16 + ln;
      out[(size_t)(n0 + rstrip + r + 8 * kh) * GD + c] = oacc[t][r];
    }
  }
}

// -----------------------------------------------------------------------------
extern "C" void kernel_launch(void* const* d_in, const int* in_sizes, int n_in,
                              void* d_out, int out_size, void* d_ws, size_t ws_size,
                              hipStream_t stream) {
  (void)in_sizes; (void)n_in; (void)out_size; (void)ws_size;
  const float* x  = (const float*)d_in[0];
  const float* Wq = (const float*)d_in[1];
  const float* bq = (const float*)d_in[2];
  const float* Wk = (const float*)d_in[3];
  const float* bk = (const float*)d_in[4];
  const float* Wv = (const float*)d_in[5];
  const float* bv = (const float*)d_in[6];
  float* out = (float*)d_out;

  char* w = (char*)d_ws;
  __bf16* Qb  = (__bf16*)w;  w += (size_t)GN * HD * 2;
  __bf16* Kb  = (__bf16*)w;  w += (size_t)GN * HD * 2;
  __bf16* Vb  = (__bf16*)w;  w += (size_t)GN * HD * 2;
  __bf16* Ktp = (__bf16*)w;  w += (size_t)GN * HD * 2;
  __bf16* Vtp = (__bf16*)w;  w += (size_t)GN * HD * 2;
  __bf16* xb  = (__bf16*)w;  w += (size_t)GN * INC * 2;
  __bf16* Wbp = (__bf16*)w;  w += (size_t)3 * INC * HD * 2;
  __bf16* kvp = (__bf16*)w;  w += (size_t)GH * GD * GD * 2;
  // accumulators: contiguous so a single zero pass covers them
  float* sqq  = (float*)w;   w += (size_t)GN * GH * 4;
  float* sqk  = (float*)w;   w += (size_t)GN * GH * 4;
  float* kv   = (float*)w;   w += (size_t)GH * GD * GD * 4;
  float* ksum = (float*)w;   w += (size_t)HD * 4;
  float* vsum = (float*)w;   w += (size_t)HD * 4;

  const int zn = GN * GH * 2 + GH * GD * GD + HD * 2;
  hipLaunchKernelGGL(k_zero, dim3((zn + 255) / 256), dim3(256), 0, stream, sqq, zn);
  hipLaunchKernelGGL(k_cvt, dim3((GN * INC) / 256), dim3(256), 0, stream, x, xb);
  hipLaunchKernelGGL(k_cvt_w_pack, dim3(3 * 65536 / 256), dim3(256), 0, stream,
                     Wq, Wk, Wv, Wbp);
  hipLaunchKernelGGL(k_proj, dim3(48, GN / 64), dim3(128), 0, stream,
                     xb, Wbp, bq, bk, bv, Qb, Kb, Vb, sqq, sqk);
  hipLaunchKernelGGL(k_norm, dim3((GN * HD) / 256), dim3(256), 0, stream,
                     Qb, Kb, sqq, sqk);
  hipLaunchKernelGGL(k_pack, dim3(GN / 64, 16), dim3(256), 0, stream, Kb, Ktp);
  hipLaunchKernelGGL(k_pack, dim3(GN / 64, 16), dim3(256), 0, stream, Vb, Vtp);
  hipLaunchKernelGGL(k_kv, dim3(16, 16), dim3(256), 0, stream, Ktp, Vtp, kv);
  hipLaunchKernelGGL(k_cvt_kv_pack, dim3(32768 / 256), dim3(256), 0, stream, kv, kvp);
  hipLaunchKernelGGL(k_sums, dim3(GN / 64), dim3(256), 0, stream, Kb, Vb, ksum, vsum);
  hipLaunchKernelGGL(k_out, dim3(GN / 64), dim3(256), 0, stream, Qb, kvp, ksum, vsum, out);
}
